// DisjointModel_47356309405950
// MI455X (gfx1250) — matrix operand order repaired and checked by reading, not compiled, not run on hardware
//
#include <hip/hip_runtime.h>
#include <hip/hip_bf16.h>
#include <math.h>

#define T_STEPS 64
#define NNODES  2048
#define NEDGES  32768
#define NHEADS  6
#define HID     32
#define NMID    16
#define NCLS    3

typedef __attribute__((ext_vector_type(16))) __bf16 v16bf;
typedef __attribute__((ext_vector_type(8)))  __bf16 v8bf;
typedef __attribute__((ext_vector_type(8)))  float  v8f;

__device__ __forceinline__ float atomic_add_f32(float* p, float v) {
  return __hip_atomic_fetch_add(p, v, __ATOMIC_RELAXED, __HIP_MEMORY_SCOPE_AGENT);
}
// float atomic max via int punning (works with -inf init)
__device__ __forceinline__ void atomic_max_f32(float* addr, float val) {
  if (val >= 0.0f) atomicMax((int*)addr, __float_as_int(val));
  else             atomicMin((unsigned int*)addr, (unsigned int)__float_as_int(val));
}

// ---------------------------------------------------------------- utilities
__global__ void fill_f32(float* __restrict__ p, float v, int n) {
  int i = blockIdx.x * blockDim.x + threadIdx.x;
  if (i < n) p[i] = v;
}

// W [fin, H, fout] f32  ->  Wb column-major [ncols, Kpad] bf16 (K zero-padded)
__global__ void wprep_kernel(const float* __restrict__ W, __bf16* __restrict__ Wb,
                             int fin, int Kpad, int ncols) {
  int i = blockIdx.x * blockDim.x + threadIdx.x;
  if (i >= ncols * Kpad) return;
  int c = i / Kpad, k = i - c * Kpad;
  Wb[i] = (k < fin) ? (__bf16)W[(size_t)k * ncols + c] : (__bf16)0.0f;
}

// x [M, fin] f32 -> A [M, Kpad] bf16 (K zero-padded)
__global__ void aprep_kernel(const float* __restrict__ x, __bf16* __restrict__ A,
                             int M, int fin, int Kpad) {
  int i = blockIdx.x * blockDim.x + threadIdx.x;
  if (i >= M * Kpad) return;
  int n = i / Kpad, k = i - n * Kpad;
  A[i] = (k < fin) ? (__bf16)x[(size_t)n * fin + k] : (__bf16)0.0f;
}

// ------------------------------------------------------------ WMMA GEMM
// C[M,NC] = A[M,K](bf16, row-major) * B(bf16, col-major [NC][K]) ; K % 32 == 0
// 2x2 register blocking: each wave computes a 32x32 block of C (4 WMMA tiles
// per k-step from 2 A-fragments + 2 B-fragments -> half the L2 traffic/WMMA).
__device__ __forceinline__ v16bf load_a_frag(const __bf16* Arow, int kb) {
  v8bf lo = *(const v8bf*)(Arow + kb);        // K = kb .. kb+7
  v8bf hi = *(const v8bf*)(Arow + kb + 16);   // K = kb+16 .. kb+23
  return __builtin_shufflevector(lo, hi, 0,1,2,3,4,5,6,7,8,9,10,11,12,13,14,15);
}

__global__ void __launch_bounds__(256)
gemm_wmma_bf16(const __bf16* __restrict__ A, const __bf16* __restrict__ B,
               float* __restrict__ C, int M, int K, int NC) {
  int tile    = blockIdx.x * blockDim.y + threadIdx.y;
  int ntiles2 = NC >> 5;                 // 32-wide column blocks
  int mt = tile / ntiles2;
  int nt = tile - mt * ntiles2;
  if (mt * 32 >= M) return;
  int lane = threadIdx.x;
  int l16  = lane & 15;
  int khiA = (lane & 16) ? 8  : 0;       // A: lanes16-31 hold K+8 block (ISA 7.12.2)
  int khiB = (lane & 16) ? 16 : 0;       // unused: B loads use kb (khiA), same K order as A
  const __bf16* Arow0 = A + (size_t)(mt * 32 + l16) * K;
  const __bf16* Arow1 = Arow0 + (size_t)16 * K;
  const __bf16* Bcol0 = B + (size_t)(nt * 32 + l16) * K;
  const __bf16* Bcol1 = Bcol0 + (size_t)16 * K;
  v8f acc00 = {}, acc01 = {}, acc10 = {}, acc11 = {};
  for (int kk = 0; kk < K; kk += 32) {
    int kb = kk + khiA;
    v16bf a0 = load_a_frag(Arow0, kb);
    v16bf a1 = load_a_frag(Arow1, kb);
    v16bf b0 = load_a_frag(Bcol0, kb);
    v16bf b1 = load_a_frag(Bcol1, kb);
    acc00 = __builtin_amdgcn_wmma_f32_16x16x32_bf16(false, a0, false, b0, (short)0, acc00, false, false);
    acc01 = __builtin_amdgcn_wmma_f32_16x16x32_bf16(false, a0, false, b1, (short)0, acc01, false, false);
    acc10 = __builtin_amdgcn_wmma_f32_16x16x32_bf16(false, a1, false, b0, (short)0, acc10, false, false);
    acc11 = __builtin_amdgcn_wmma_f32_16x16x32_bf16(false, a1, false, b1, (short)0, acc11, false, false);
  }
  int mo = (lane & 16) ? 8 : 0;          // D: VGPR v -> M = v + (lane>=16 ? 8 : 0)
  float* C00 = C + (size_t)(mt * 32) * NC + nt * 32 + l16;
  float* C10 = C00 + (size_t)16 * NC;
#pragma unroll
  for (int v = 0; v < 8; v++) {
    size_t r = (size_t)(v + mo) * NC;
    C00[r]      = acc00[v];
    C00[r + 16] = acc01[v];
    C10[r]      = acc10[v];
    C10[r + 16] = acc11[v];
  }
}

// ------------------------------------------------------------ attention
// es[n,h] = sum_o hW[n,h,o]*aS[h,o] ; ed likewise
__global__ void esed_kernel(const float* __restrict__ hW, const float* __restrict__ aS,
                            const float* __restrict__ aD, float* __restrict__ es,
                            float* __restrict__ ed, int M, int fout) {
  int idx = blockIdx.x * blockDim.x + threadIdx.x;
  if (idx >= M * NHEADS) return;
  int n = idx / NHEADS, h = idx - n * NHEADS;
  const float* base = hW + (size_t)n * NHEADS * fout + h * fout;
  const float* s = aS + h * fout;
  const float* d = aD + h * fout;
  float accS = 0.f, accD = 0.f;
  for (int o = 0; o < fout; o++) { float v = base[o]; accS += v * s[o]; accD += v * d[o]; }
  es[idx] = accS; ed[idx] = accD;
}

__global__ void edge_max_kernel(const int* __restrict__ ei, const float* __restrict__ es,
                                const float* __restrict__ ed, float* __restrict__ ebuf,
                                float* __restrict__ m, int tc, int t0) {
  int idx = blockIdx.x * blockDim.x + threadIdx.x;
  int total = tc * NEDGES * NHEADS;
  if (idx >= total) return;
  int tl = idx / (NEDGES * NHEADS);
  int r  = idx - tl * (NEDGES * NHEADS);
  int e  = r / NHEADS, h = r - e * NHEADS;
  const int* eit = ei + (size_t)(t0 + tl) * 2 * NEDGES;
  int src = eit[e], dst = eit[NEDGES + e];
  float v = es[((size_t)tl * NNODES + src) * NHEADS + h]
          + ed[((size_t)tl * NNODES + dst) * NHEADS + h];
  v = (v > 0.0f) ? v : 0.2f * v;                       // leaky_relu 0.2
  ebuf[idx] = v;
  atomic_max_f32(&m[((size_t)tl * NNODES + dst) * NHEADS + h], v);
}

__global__ void edge_exp_kernel(const int* __restrict__ ei, float* __restrict__ ebuf,
                                const float* __restrict__ m, float* __restrict__ den,
                                int tc, int t0) {
  int idx = blockIdx.x * blockDim.x + threadIdx.x;
  int total = tc * NEDGES * NHEADS;
  if (idx >= total) return;
  int tl = idx / (NEDGES * NHEADS);
  int r  = idx - tl * (NEDGES * NHEADS);
  int e  = r / NHEADS, h = r - e * NHEADS;
  const int* eit = ei + (size_t)(t0 + tl) * 2 * NEDGES;
  int dst = eit[NEDGES + e];
  size_t di = ((size_t)tl * NNODES + dst) * NHEADS + h;
  float ex = __expf(ebuf[idx] - m[di]);
  ebuf[idx] = ex;
  atomic_add_f32(&den[di], ex);
}

__global__ void edge_coef_kernel(const int* __restrict__ ei, const float* __restrict__ ew,
                                 float* __restrict__ ebuf, const float* __restrict__ den,
                                 int tc, int t0) {
  int idx = blockIdx.x * blockDim.x + threadIdx.x;
  int total = tc * NEDGES * NHEADS;
  if (idx >= total) return;
  int tl = idx / (NEDGES * NHEADS);
  int r  = idx - tl * (NEDGES * NHEADS);
  int e  = r / NHEADS, h = r - e * NHEADS;
  const int* eit = ei + (size_t)(t0 + tl) * 2 * NEDGES;
  int dst = eit[NEDGES + e];
  size_t di = ((size_t)tl * NNODES + dst) * NHEADS + h;
  ebuf[idx] = ebuf[idx] * ew[(size_t)(t0 + tl) * NEDGES + e] / (den[di] + 1e-16f);
}

__global__ void scatter_kernel(const int* __restrict__ ei, const float* __restrict__ ebuf,
                               const float* __restrict__ hW, float* __restrict__ accb,
                               int tc, int t0, int fout) {
  int idx = blockIdx.x * blockDim.x + threadIdx.x;
  int total = tc * NEDGES * NHEADS;
  if (idx >= total) return;
  int tl = idx / (NEDGES * NHEADS);
  int r  = idx - tl * (NEDGES * NHEADS);
  int e  = r / NHEADS, h = r - e * NHEADS;
  const int* eit = ei + (size_t)(t0 + tl) * 2 * NEDGES;
  int src = eit[e], dst = eit[NEDGES + e];
  float c = ebuf[idx];
  const float* hs = hW   + ((size_t)tl * NNODES + src) * (NHEADS * fout) + h * fout;
  float*       ad = accb + ((size_t)tl * NNODES + dst) * (NHEADS * fout) + h * fout;
#pragma unroll 4
  for (int o = 0; o < fout; o++) atomic_add_f32(&ad[o], c * hs[o]);
}

// mean over heads + bias (+elu); write bf16 input for next layer, or f32 for readout
__global__ void finish_kernel(const float* __restrict__ accb, const float* __restrict__ bias,
                              float* __restrict__ hout, __bf16* __restrict__ Anext,
                              int M, int fout, int act) {
  int i = blockIdx.x * blockDim.x + threadIdx.x;
  if (i >= M * fout) return;
  int n = i / fout, o = i - n * fout;
  const float* a = accb + (size_t)n * NHEADS * fout + o;
  float s = 0.f;
#pragma unroll
  for (int h = 0; h < NHEADS; h++) s += a[(size_t)h * fout];
  s = s * (1.0f / NHEADS) + bias[o];
  if (act) s = (s > 0.0f) ? s : (__expf(s) - 1.0f);      // elu
  if (Anext) Anext[(size_t)n * fout + o] = (__bf16)s;
  if (hout)  hout[(size_t)n * fout + o] = s;
}

// g[t, goff+o] = sum_n xn[t,n] * h[tl*N+n, o]   (block per tl, 64 threads = o)
__global__ void readout_kernel(const float* __restrict__ xn, const float* __restrict__ h,
                               float* __restrict__ g, int t0, int goff) {
  int tl = blockIdx.x;
  int o  = threadIdx.x;
  const float* xr = xn + (size_t)(t0 + tl) * NNODES;
  const float* hr = h  + (size_t)tl * NNODES * 64 + o;
  float s = 0.f;
  for (int n = 0; n < NNODES; n++) s += xr[n] * hr[(size_t)n * 64];
  g[(size_t)(t0 + tl) * 128 + goff + o] = s;
}

// ------------------------------------------------------------ GRU + MLP tail
__global__ void __launch_bounds__(96)
gru_mlp_kernel(const float* __restrict__ g, const float* __restrict__ Wih,
               const float* __restrict__ Whh, const float* __restrict__ bih,
               const float* __restrict__ bhh, const float* __restrict__ Wc1,
               const float* __restrict__ bc1, const float* __restrict__ Wc2,
               const float* __restrict__ bc2, float* __restrict__ out) {
  __shared__ float h[HID];
  __shared__ float gi[3 * HID], gh[3 * HID];
  __shared__ float rnn[T_STEPS * HID];
  int tid = threadIdx.x;
  if (tid < HID) h[tid] = 0.f;
  __syncthreads();
  for (int t = 0; t < T_STEPS; t++) {
    const float* x = g + (size_t)t * 128;
    float a = bih[tid];
    for (int k = 0; k < 128; k++) a += Wih[(size_t)tid * 128 + k] * x[k];
    float b = bhh[tid];
    for (int k = 0; k < HID; k++) b += Whh[(size_t)tid * HID + k] * h[k];
    gi[tid] = a; gh[tid] = b;
    __syncthreads();
    if (tid < HID) {
      float r  = 1.f / (1.f + __expf(-(gi[tid] + gh[tid])));
      float z  = 1.f / (1.f + __expf(-(gi[HID + tid] + gh[HID + tid])));
      float nn = tanhf(gi[2 * HID + tid] + r * gh[2 * HID + tid]);
      float h2 = (1.f - z) * nn + z * h[tid];
      rnn[t * HID + tid] = h2;
      h[tid] = h2;
    }
    __syncthreads();
  }
  for (int t = tid; t < T_STEPS; t += 96) {
    float hid[NMID];
    for (int j = 0; j < NMID; j++) {
      float s = bc1[j];
      for (int k = 0; k < HID; k++) s += rnn[t * HID + k] * Wc1[k * NMID + j];
      hid[j] = s > 0.f ? s : 0.f;
    }
    for (int c = 0; c < NCLS; c++) {
      float s = bc2[c];
      for (int j = 0; j < NMID; j++) s += hid[j] * Wc2[j * NCLS + c];
      out[t * NCLS + c] = s;
    }
  }
}

// ------------------------------------------------------------ host driver
extern "C" void kernel_launch(void* const* d_in, const int* in_sizes, int n_in,
                              void* d_out, int out_size, void* d_ws, size_t ws_size,
                              hipStream_t stream) {
  (void)in_sizes; (void)n_in; (void)out_size;
  const float* x[2]  = {(const float*)d_in[0], (const float*)d_in[1]};
  const int*   ei[2] = {(const int*)d_in[2], (const int*)d_in[3]};
  const float* ew[2] = {(const float*)d_in[4], (const float*)d_in[5]};
  const float* xn[2] = {(const float*)d_in[6], (const float*)d_in[7]};
  const float *W[4], *aS[4], *aD[4], *bg[4];
  for (int i = 0; i < 4; i++) {
    W[i]  = (const float*)d_in[9 + 4 * i];
    aS[i] = (const float*)d_in[10 + 4 * i];
    aD[i] = (const float*)d_in[11 + 4 * i];
    bg[i] = (const float*)d_in[12 + 4 * i];
  }
  const float* Wih = (const float*)d_in[25];
  const float* Whh = (const float*)d_in[26];
  const float* bih = (const float*)d_in[27];
  const float* bhh = (const float*)d_in[28];
  const float* Wc1 = (const float*)d_in[29];
  const float* bc1 = (const float*)d_in[30];
  const float* Wc2 = (const float*)d_in[31];
  const float* bc2 = (const float*)d_in[32];
  float* out = (float*)d_out;

  const int fin_[4]  = {4, 128, 128, 64};
  const int kpad_[4] = {32, 128, 128, 64};
  const int fout_[4] = {128, 128, 64, 64};

  size_t off = 0;
  auto walloc = [&](size_t bytes) -> void* {
    void* p = (char*)d_ws + off;
    off += (bytes + 255) & ~(size_t)255;
    return p;
  };
  __bf16* Wb[4];
  for (int i = 0; i < 4; i++)
    Wb[i] = (__bf16*)walloc((size_t)kpad_[i] * NHEADS * fout_[i] * 2);
  float* gbuf = (float*)walloc((size_t)T_STEPS * 128 * 4);
  size_t fixed = off;

  const size_t perT = (size_t)NNODES * 128 * 2        // A (bf16)
                    + (size_t)NNODES * 768 * 4        // hW
                    + (size_t)NNODES * 768 * 4        // acc
                    + (size_t)NNODES * NHEADS * 4 * 4 // es, ed, m, den
                    + (size_t)NEDGES * NHEADS * 4     // edge buffer
                    + (size_t)NNODES * 64 * 4         // hout
                    + 10 * 256;                       // align slack
  int TC = 1;
  if (ws_size > fixed + perT) {
    size_t t = (ws_size - fixed) / perT;
    TC = (t > (size_t)T_STEPS) ? T_STEPS : (int)t;
    if (TC < 1) TC = 1;
  }
  __bf16* Abuf  = (__bf16*)walloc((size_t)TC * NNODES * 128 * 2);
  float*  hWbuf = (float*)walloc((size_t)TC * NNODES * 768 * 4);
  float*  accb  = (float*)walloc((size_t)TC * NNODES * 768 * 4);
  float*  esb   = (float*)walloc((size_t)TC * NNODES * NHEADS * 4);
  float*  edb   = (float*)walloc((size_t)TC * NNODES * NHEADS * 4);
  float*  mb    = (float*)walloc((size_t)TC * NNODES * NHEADS * 4);
  float*  denb  = (float*)walloc((size_t)TC * NNODES * NHEADS * 4);
  float*  ebuf  = (float*)walloc((size_t)TC * NEDGES * NHEADS * 4);
  float*  houtb = (float*)walloc((size_t)TC * NNODES * 64 * 4);

  for (int i = 0; i < 4; i++) {
    int n = kpad_[i] * NHEADS * fout_[i];
    wprep_kernel<<<(n + 255) / 256, 256, 0, stream>>>(W[i], Wb[i], fin_[i], kpad_[i],
                                                      NHEADS * fout_[i]);
  }

  for (int t0 = 0; t0 < T_STEPS; t0 += TC) {
    int tc = T_STEPS - t0 < TC ? T_STEPS - t0 : TC;
    int M  = tc * NNODES;
    for (int gI = 0; gI < 2; gI++) {
      {
        int n = M * kpad_[0];
        aprep_kernel<<<(n + 255) / 256, 256, 0, stream>>>(
            x[gI] + (size_t)t0 * NNODES * 4, Abuf, M, 4, kpad_[0]);
      }
      for (int L = 0; L < 4; L++) {
        int fo = fout_[L], nc = NHEADS * fo, K = kpad_[L];
        fill_f32<<<((M * nc) + 255) / 256, 256, 0, stream>>>(accb, 0.f, M * nc);
        fill_f32<<<((M * NHEADS) + 255) / 256, 256, 0, stream>>>(mb, -__builtin_huge_valf(), M * NHEADS);
        fill_f32<<<((M * NHEADS) + 255) / 256, 256, 0, stream>>>(denb, 0.f, M * NHEADS);
        {
          int tiles = (M / 32) * (nc / 32);     // 2x2-blocked tiles
          dim3 blk(32, 8);
          gemm_wmma_bf16<<<(tiles + 7) / 8, blk, 0, stream>>>(Abuf, Wb[L], hWbuf, M, K, nc);
        }
        esed_kernel<<<((M * NHEADS) + 255) / 256, 256, 0, stream>>>(hWbuf, aS[L], aD[L],
                                                                    esb, edb, M, fo);
        int ne = tc * NEDGES * NHEADS;
        edge_max_kernel<<<(ne + 255) / 256, 256, 0, stream>>>(ei[gI], esb, edb, ebuf, mb, tc, t0);
        edge_exp_kernel<<<(ne + 255) / 256, 256, 0, stream>>>(ei[gI], ebuf, mb, denb, tc, t0);
        edge_coef_kernel<<<(ne + 255) / 256, 256, 0, stream>>>(ei[gI], ew[gI], ebuf, denb, tc, t0);
        scatter_kernel<<<(ne + 255) / 256, 256, 0, stream>>>(ei[gI], ebuf, hWbuf, accb, tc, t0, fo);
        int nf = M * fo;
        finish_kernel<<<(nf + 255) / 256, 256, 0, stream>>>(
            accb, bg[L], (L == 3) ? houtb : (float*)nullptr,
            (L < 3) ? Abuf : (__bf16*)nullptr, M, fo, (L < 3) ? 1 : 0);
      }
      readout_kernel<<<tc, 64, 0, stream>>>(xn[gI], houtb, gbuf, t0, gI * 64);
    }
  }
  gru_mlp_kernel<<<1, 96, 0, stream>>>(gbuf, Wih, Whh, bih, bhh, Wc1, bc1, Wc2, bc2, out);
}
